// DynamicHippocampus_45243185496633
// MI455X (gfx1250) — compile-verified, matrix operand order checked by reading, not run on hardware
//
#include <hip/hip_runtime.h>

// ---------------- problem constants (match reference) ----------------
#define N_EC   100000
#define N_DG   400000
#define N_CA3  120000
#define N_CA1  100000
#define PP_NNZ 10000000
#define MF_NNZ 8000000
#define RC_NNZ 6000000
#define SC_NNZ 6000000
#define T_STEPS 4

#define IZH_A   0.02f
#define IZH_B   0.2f
#define TAU_I   0.9f
#define THR_I   1.0f
#define INH_G   2.0f

typedef __attribute__((ext_vector_type(4)))  int      v4i;
typedef __attribute__((ext_vector_type(4)))  float    v4f;
typedef __attribute__((ext_vector_type(16))) _Float16 v16h;
typedef __attribute__((ext_vector_type(8)))  float    v8f;

// ---------------- simple fill ----------------
__global__ void fill_kernel(float* __restrict__ p, float v, int n) {
    int i = blockIdx.x * blockDim.x + threadIdx.x;
    if (i < n) p[i] = v;
}

// ---------------- Izhikevich core (matches reference exactly) ----------------
__device__ __forceinline__ void izh_core(float& v, float& u, float c, float d,
                                         float I, float& spike) {
    float nv = v + (0.04f * v * v + 5.0f * v + 140.0f - u + I) * 0.5f;
    nv = fminf(fmaxf(nv, -90.0f), 40.0f);              // clip first (as in jax ref)
    float nu = u + IZH_A * (IZH_B * nv - u) * 0.5f;    // du uses clamped post-update v
    spike = (nv >= 30.0f) ? 1.0f : 0.0f;
    v = (spike > 0.0f) ? c : nv;
    u = (spike > 0.0f) ? (nu + d) : nu;
}

__global__ void izh_ec_kernel(float* __restrict__ v, float* __restrict__ u,
                              float* __restrict__ s,
                              const float* __restrict__ c, const float* __restrict__ d,
                              const float* __restrict__ drive, int n) {
    int i = blockIdx.x * blockDim.x + threadIdx.x;
    if (i >= n) return;
    float vv = v[i], uu = u[i], sp;
    izh_core(vv, uu, c[i], d[i], drive[i], sp);
    v[i] = vv; u[i] = uu; s[i] = sp;
}

__global__ void izh_pop_kernel(float* __restrict__ v, float* __restrict__ u,
                               float* __restrict__ s,      // may be nullptr (CA1)
                               const float* __restrict__ c, const float* __restrict__ d,
                               const float* __restrict__ Iarr,
                               const float* __restrict__ inh, int n) {
    int i = blockIdx.x * blockDim.x + threadIdx.x;
    if (i >= n) return;
    float vv = v[i], uu = u[i], sp;
    izh_core(vv, uu, c[i], d[i], Iarr[i] - inh[0], sp);
    v[i] = vv; u[i] = uu;
    if (s) s[i] = sp;
}

// ---------------- sparse transmit: 4 edges/thread, NT 128-bit streams ----------------
// Dominant cost of the whole model: 30M edges/step * 12B streamed once (NT hint keeps
// them out of the L2 working set); spike gathers + f32 atomics resolve in L2.
__global__ void transmit_kernel(const v4i* __restrict__ src, const v4i* __restrict__ tgt,
                                const v4f* __restrict__ val, const float* __restrict__ spk,
                                float* __restrict__ acc, int nquads) {
    int i = blockIdx.x * blockDim.x + threadIdx.x;
    if (i >= nquads) return;
    v4i s4 = __builtin_nontemporal_load(&src[i]);
    v4i t4 = __builtin_nontemporal_load(&tgt[i]);
    v4f w4 = __builtin_nontemporal_load(&val[i]);
#pragma unroll
    for (int j = 0; j < 4; ++j) {
        if (spk[s4[j]] > 0.0f)   // spikes are 0/1, so val*spk == val when firing
            __hip_atomic_fetch_add(&acc[t4[j]], w4[j],
                                   __ATOMIC_RELAXED, __HIP_MEMORY_SCOPE_AGENT);
    }
}

// ---------------- mean reduction; final 256->1 collapse via WMMA ones-trick ----------
// Each block: vectorized grid-stride f32 accumulate -> 256 partials scaled by 1/N
// (small, so f16 conversion is safe) stored in a zero-padded 512-slot LDS region ->
// wave 0 loads one contiguous v16h per lane (no divergence) and performs
// D = A x ones with v_wmma_f32_16x16x32_f16 (row sums). Lanes 0 and 16 hold
// D[0..7][0] / D[8..15][0]; both add atomically (backend folds to one atomic).
__global__ void reduce_mean_wmma_kernel(const v4f* __restrict__ x4, int nquads,
                                        float inv_n, float* __restrict__ out_sum) {
    __shared__ _Float16 parts[512];
    float acc = 0.0f;
    for (int i = blockIdx.x * blockDim.x + threadIdx.x; i < nquads;
         i += gridDim.x * blockDim.x) {
        v4f q = x4[i];
        acc += (q[0] + q[1]) + (q[2] + q[3]);
    }
    parts[threadIdx.x]       = (_Float16)(acc * inv_n);
    parts[threadIdx.x + 256] = (_Float16)0.0f;         // zero pad: upper half of A
    __syncthreads();

    if (threadIdx.x < 32) {              // whole wave 0 enters: EXEC all ones (WMMA req.)
        int lane = threadIdx.x;
        v16h a = ((const v16h*)parts)[lane];           // contiguous 32B per lane
        v16h b;
#pragma unroll
        for (int k = 0; k < 16; ++k) b[k] = (_Float16)1.0f;
        v8f cacc = {};
        // D[m][n] = sum_k A[m][k] * 1  (+0): row sums of the 512-slot A tile
        cacc = __builtin_amdgcn_wmma_f32_16x16x32_f16(
            /*neg_a=*/false, a, /*neg_b=*/false, b,
            /*c_mod=*/(short)0, cacc, /*reuse_a=*/false, /*reuse_b=*/false);
        float t = (cacc[0] + cacc[1]) + (cacc[2] + cacc[3]) +
                  (cacc[4] + cacc[5]) + (cacc[6] + cacc[7]);
        if ((lane & 15) == 0)            // lanes 0 and 16: the two column-0 halves
            __hip_atomic_fetch_add(out_sum, t,
                                   __ATOMIC_RELAXED, __HIP_MEMORY_SCOPE_AGENT);
    }
}

// ---------------- scalar LIF inhibitory population (all neurons identical) ----------
__global__ void lif_kernel(const float* __restrict__ mean_ptr,
                           float* __restrict__ iv, float* __restrict__ inh) {
    float v = TAU_I * iv[0] + (1.0f - TAU_I) * mean_ptr[0];
    float s = (v >= THR_I) ? 1.0f : 0.0f;
    iv[0]  = (s > 0.0f) ? 0.0f : v;
    inh[0] = INH_G * s;                  // INH_GAIN * is.mean()
}

// ---------------- host-side orchestration ----------------
static inline int gridFor(int n, int b) { return (n + b - 1) / b; }

extern "C" void kernel_launch(void* const* d_in, const int* in_sizes, int n_in,
                              void* d_out, int out_size, void* d_ws, size_t ws_size,
                              hipStream_t stream) {
    (void)in_sizes; (void)n_in; (void)out_size; (void)ws_size;
    const float* drive  = (const float*)d_in[0];
    const int*   pp_src = (const int*)d_in[1];
    const int*   pp_tgt = (const int*)d_in[2];
    const float* pp_val = (const float*)d_in[3];
    const int*   mf_src = (const int*)d_in[4];
    const int*   mf_tgt = (const int*)d_in[5];
    const float* mf_val = (const float*)d_in[6];
    const int*   rc_src = (const int*)d_in[7];
    const int*   rc_tgt = (const int*)d_in[8];
    const float* rc_val = (const float*)d_in[9];
    const int*   sc_src = (const int*)d_in[10];
    const int*   sc_tgt = (const int*)d_in[11];
    const float* sc_val = (const float*)d_in[12];
    const float* ec_c = (const float*)d_in[13], *ec_d = (const float*)d_in[14];
    const float* dg_c = (const float*)d_in[15], *dg_d = (const float*)d_in[16];
    const float* c3_c = (const float*)d_in[17], *c3_d = (const float*)d_in[18];
    const float* c1_c = (const float*)d_in[19], *c1_d = (const float*)d_in[20];

    // workspace layout (floats)
    float* ws = (float*)d_ws;
    float* ec_v = ws;              float* ec_u = ec_v + N_EC;  float* ec_s = ec_u + N_EC;
    float* dg_v = ec_s + N_EC;     float* dg_u = dg_v + N_DG;
    float* dg_s = dg_u + N_DG;     float* dg_I = dg_s + N_DG;
    float* c3_v = dg_I + N_DG;     float* c3_u = c3_v + N_CA3;
    float* c3_s = c3_u + N_CA3;    float* c3_I = c3_s + N_CA3;
    float* c1_u = c3_I + N_CA3;    float* c1_I = c1_u + N_CA1;
    float* iv   = c1_I + N_CA1;    // iv[0..2] = iv_dg, iv_c3, iv_c1
    float* sums = iv + 3;          // sums[0..2] = sum_dg, sum_c3, sum_c1
    float* inh  = sums + 3;        // inh[0..2] = inh_dg, inh_c3, inh_c1
    float* c1_v = (float*)d_out;   // final CA1 membrane potential IS the output

    const int B  = 256;
    const int RB = 256;            // reduction blocks
    #define FILL(p, v, n) fill_kernel<<<gridFor((n), B), B, 0, stream>>>((p), (v), (n))

    // ---- initial state (every call: deterministic, graph-capture safe) ----
    FILL(ec_v, -65.0f, N_EC);  FILL(ec_u, IZH_B * -65.0f, N_EC);
    FILL(dg_v, -65.0f, N_DG);  FILL(dg_u, IZH_B * -65.0f, N_DG);
    FILL(c3_v, -65.0f, N_CA3); FILL(c3_u, IZH_B * -65.0f, N_CA3);
    FILL(c1_v, -65.0f, N_CA1); FILL(c1_u, IZH_B * -65.0f, N_CA1);
    FILL(c3_s, 0.0f, N_CA3);
    FILL(iv,   0.0f, 3);

    for (int t = 0; t < T_STEPS; ++t) {
        FILL(dg_I, 0.0f, N_DG);
        FILL(c3_I, 0.0f, N_CA3);
        FILL(c1_I, 0.0f, N_CA1);
        FILL(sums, 0.0f, 3);

        // EC -> perforant path -> DG
        izh_ec_kernel<<<gridFor(N_EC, B), B, 0, stream>>>(
            ec_v, ec_u, ec_s, ec_c, ec_d, drive + (size_t)t * N_EC, N_EC);
        transmit_kernel<<<gridFor(PP_NNZ / 4, B), B, 0, stream>>>(
            (const v4i*)pp_src, (const v4i*)pp_tgt, (const v4f*)pp_val, ec_s, dg_I, PP_NNZ / 4);
        reduce_mean_wmma_kernel<<<RB, B, 0, stream>>>(
            (const v4f*)dg_I, N_DG / 4, 1.0f / N_DG, &sums[0]);
        lif_kernel<<<1, 1, 0, stream>>>(&sums[0], &iv[0], &inh[0]);
        izh_pop_kernel<<<gridFor(N_DG, B), B, 0, stream>>>(
            dg_v, dg_u, dg_s, dg_c, dg_d, dg_I, &inh[0], N_DG);

        // DG -> mossy fibers + CA3 recurrent (uses previous-step c3_s) -> CA3
        transmit_kernel<<<gridFor(MF_NNZ / 4, B), B, 0, stream>>>(
            (const v4i*)mf_src, (const v4i*)mf_tgt, (const v4f*)mf_val, dg_s, c3_I, MF_NNZ / 4);
        transmit_kernel<<<gridFor(RC_NNZ / 4, B), B, 0, stream>>>(
            (const v4i*)rc_src, (const v4i*)rc_tgt, (const v4f*)rc_val, c3_s, c3_I, RC_NNZ / 4);
        reduce_mean_wmma_kernel<<<RB, B, 0, stream>>>(
            (const v4f*)c3_I, N_CA3 / 4, 1.0f / N_CA3, &sums[1]);
        lif_kernel<<<1, 1, 0, stream>>>(&sums[1], &iv[1], &inh[1]);
        izh_pop_kernel<<<gridFor(N_CA3, B), B, 0, stream>>>(
            c3_v, c3_u, c3_s, c3_c, c3_d, c3_I, &inh[1], N_CA3);  // overwrites c3_s AFTER rc use

        // CA3 -> Schaffer collaterals (uses new c3_s) -> CA1
        transmit_kernel<<<gridFor(SC_NNZ / 4, B), B, 0, stream>>>(
            (const v4i*)sc_src, (const v4i*)sc_tgt, (const v4f*)sc_val, c3_s, c1_I, SC_NNZ / 4);
        reduce_mean_wmma_kernel<<<RB, B, 0, stream>>>(
            (const v4f*)c1_I, N_CA1 / 4, 1.0f / N_CA1, &sums[2]);
        lif_kernel<<<1, 1, 0, stream>>>(&sums[2], &iv[2], &inh[2]);
        izh_pop_kernel<<<gridFor(N_CA1, B), B, 0, stream>>>(
            c1_v, c1_u, nullptr, c1_c, c1_d, c1_I, &inh[2], N_CA1);
    }
    #undef FILL
}
